// GraphFeatEncoder_29652454211889
// MI455X (gfx1250) — matrix operand order repaired and checked by reading, not compiled, not run on hardware
//
#include <hip/hip_runtime.h>
#include <math.h>

typedef __attribute__((ext_vector_type(16))) __bf16 v16bf;
typedef __attribute__((ext_vector_type(8)))  __bf16 bf16x8;
typedef __attribute__((ext_vector_type(8)))  float  v8f;

#define E_EDGES 40000
#define N_NODES 20000
#define H 256

// ---------------------------------------------------------------------------
// WMMA helpers (CDNA5 v_wmma_f32_16x16x32_bf16)
// A fragment: row-major source, lane L holds row M=L%16; element j (0..15):
//   K = (j<8 ? (L/16)*8 + j : 16 + (L/16)*8 + (j-8))   -> two 16B chunks
// B fragment: weight row N=L%16 (weights are [out=N, K] row-major == B^T);
//   element j: same K as A fragment                     -> two 16B chunks
// ---------------------------------------------------------------------------
__device__ __forceinline__ v16bf make_frag(const __bf16* p0, const __bf16* p1) {
  bf16x8 lo = *(const bf16x8*)p0;
  bf16x8 hi = *(const bf16x8*)p1;
  v16bf f;
#pragma unroll
  for (int i = 0; i < 8; ++i) { f[i] = lo[i]; f[i + 8] = hi[i]; }
  return f;
}

__device__ __forceinline__ v8f wmma_bf16(v16bf a, v16bf b, v8f c) {
  return __builtin_amdgcn_wmma_f32_16x16x32_bf16(false, a, false, b, (short)0, c,
                                                 false, false);
}

__device__ __forceinline__ float sigmoidf_(float x) {
  return 1.0f / (1.0f + expf(-x));
}

__device__ __forceinline__ void wait_asynccnt0() {
#if __has_builtin(__builtin_amdgcn_s_wait_asynccnt)
  __builtin_amdgcn_s_wait_asynccnt(0);
#else
  asm volatile("s_wait_asynccnt 0x0" ::: "memory");
#endif
}

// ---------------------------------------------------------------------------
// Weight conversion: f32 slice -> bf16, zero-padded to dstCols
// ---------------------------------------------------------------------------
__global__ void k_conv_w(const float* __restrict__ src, __bf16* __restrict__ dst,
                         int rows, int srcCols, int colOff, int copyCols,
                         int dstCols) {
  int i = blockIdx.x * blockDim.x + threadIdx.x;
  int total = rows * dstCols;
  if (i >= total) return;
  int r = i / dstCols, c = i % dstCols;
  float v = (c < copyCols) ? src[(size_t)r * srcCols + colOff + c] : 0.0f;
  dst[i] = (__bf16)v;
}

// x = [fnode[src] | bond feats] padded 111 -> 128, bf16
__global__ void k_build_x(const float* __restrict__ fnode,
                          const float* __restrict__ fmess,
                          __bf16* __restrict__ xb) {
  int i = blockIdx.x * blockDim.x + threadIdx.x;
  if (i >= E_EDGES * 128) return;
  int e = i >> 7, c = i & 127;
  float v;
  if (c < 98) {
    int src = (int)fmess[(size_t)e * 15];
    v = fnode[(size_t)src * 98 + c];
  } else if (c < 111) {
    v = fmess[(size_t)e * 15 + 2 + (c - 98)];
  } else {
    v = 0.0f;
  }
  xb[i] = (__bf16)v;
}

// fnode padded 98 -> 128, bf16
__global__ void k_build_fn(const float* __restrict__ fnode,
                           __bf16* __restrict__ fnb) {
  int i = blockIdx.x * blockDim.x + threadIdx.x;
  if (i >= N_NODES * 128) return;
  int n = i >> 7, c = i & 127;
  fnb[i] = (__bf16)((c < 98) ? fnode[(size_t)n * 98 + c] : 0.0f);
}

__global__ void k_zero_b(__bf16* __restrict__ p, int n) {
  int i = blockIdx.x * blockDim.x + threadIdx.x;
  if (i < n) p[i] = (__bf16)0.0f;
}

// ---------------------------------------------------------------------------
// Generic GEMM + bias: C[rows,256] = A[rows,Kpad](bf16) @ W[256,Kpad]^T + bias
// block = 256 threads (8 waves), 16 rows per block; wave w -> n-tiles {w,w+8}
// ---------------------------------------------------------------------------
__global__ __launch_bounds__(256) void k_gemm_bias(
    const __bf16* __restrict__ A, const __bf16* __restrict__ W,
    const float* __restrict__ bias, float* __restrict__ C, int Kpad) {
  int lane = threadIdx.x & 31;
  int wave = threadIdx.x >> 5;
  int half = lane >> 4, l16 = lane & 15;
  int mbase = blockIdx.x * 16;
  const __bf16* arow = A + (size_t)(mbase + l16) * Kpad;
#pragma unroll
  for (int t = 0; t < 2; ++t) {
    int nt = wave + t * 8;
    const __bf16* wrow = W + (size_t)(nt * 16 + l16) * Kpad;
    v8f acc = {};
    for (int kk = 0; kk < Kpad; kk += 32) {
      v16bf a = make_frag(arow + kk + half * 8, arow + kk + 16 + half * 8);
      v16bf b = make_frag(wrow + kk + half * 8, wrow + kk + 16 + half * 8);
      acc = wmma_bf16(a, b, acc);
    }
    int n = nt * 16 + l16;
    float bv = bias[n];
#pragma unroll
    for (int r = 0; r < 8; ++r) {
      int m = mbase + r + half * 8;
      C[(size_t)m * H + n] = acc[r] + bv;
    }
  }
}

// ---------------------------------------------------------------------------
// Gather + neighbor sum: sum_h[row] = sum_k hb[graph[row,k]]
// one wave per row; lane owns 8 contiguous columns (16B bf16 chunks)
// ---------------------------------------------------------------------------
__global__ __launch_bounds__(256) void k_gather_sum(
    const __bf16* __restrict__ hb, const int* __restrict__ graph,
    float* __restrict__ sum_f, __bf16* __restrict__ sum_b, int rows) {
  int lane = threadIdx.x & 31;
  int wave = threadIdx.x >> 5;
  int row = blockIdx.x * 8 + wave;
  if (row >= rows) return;
  float acc[8] = {};
#pragma unroll
  for (int k = 0; k < 6; ++k) {
    int idx = graph[row * 6 + k];
    bf16x8 v = *(const bf16x8*)(hb + (size_t)idx * H + lane * 8);
#pragma unroll
    for (int j = 0; j < 8; ++j) acc[j] += (float)v[j];
  }
  size_t off = (size_t)row * H + lane * 8;
  if (sum_f) {
#pragma unroll
    for (int j = 0; j < 8; ++j) sum_f[off + j] = acc[j];
  }
  bf16x8 ob;
#pragma unroll
  for (int j = 0; j < 8; ++j) ob[j] = (__bf16)acc[j];
  *(bf16x8*)(sum_b + off) = ob;
}

// ---------------------------------------------------------------------------
// Big r-GEMM, fused: for each (edge,nb) row:
//   r = sigmoid(h_nei @ U_r^T + Wr_x[e])      (U_r_b folded into Wr_x)
//   sum_gh[e] = sum_nb r * h_nei
// block = 8 edges = 48 rows x 256 cols.
// The 48 gathered h rows (24 KB) are async-staged to LDS once per block
// (global_load_async_to_lds_b128, ASYNCcnt) and reused by all 8 waves for
// both the WMMA A-fragments and the r*h epilogue. Products staged in LDS,
// fixed-order 6-way sum -> deterministic, no float atomics.
// ---------------------------------------------------------------------------
__global__ __launch_bounds__(256) void k_r_gemm(
    const __bf16* __restrict__ hb, const int* __restrict__ bgraph,
    const __bf16* __restrict__ Ur, const float* __restrict__ Wrx,
    __bf16* __restrict__ sum_gh) {
  __shared__ __bf16 ldsA[48 * 256];  // 24 KB gathered h_nei rows
  __shared__ float  ldsP[48 * 256];  // 48 KB r*h products
  int lane = threadIdx.x & 31;
  int wave = threadIdx.x >> 5;
  int half = lane >> 4, l16 = lane & 15;
  int eb = blockIdx.x;  // 8 edges per block

  // Async-stage 48 gathered rows (16B chunks, 6 per thread) into LDS.
  for (int c = threadIdx.x; c < 48 * 32; c += 256) {
    int row = c >> 5, sub = c & 31;
    int idx = bgraph[eb * 48 + row];
    const __bf16* g = hb + (size_t)idx * H + sub * 8;
    unsigned lo = (unsigned)(uintptr_t)&ldsA[row * 256 + sub * 8];
    asm volatile("global_load_async_to_lds_b128 %0, %1, off"
                 :: "v"(lo), "v"(g) : "memory");
  }
  wait_asynccnt0();
  __syncthreads();

#pragma unroll
  for (int t = 0; t < 2; ++t) {
    int nt = wave + t * 8;
    const __bf16* wrow = Ur + (size_t)(nt * 16 + l16) * H;
#pragma unroll
    for (int mt = 0; mt < 3; ++mt) {
      const __bf16* arow = &ldsA[(mt * 16 + l16) * 256];
      v8f acc = {};
      for (int kk = 0; kk < H; kk += 32) {
        v16bf a = make_frag(arow + kk + half * 8, arow + kk + 16 + half * 8);
        v16bf b = make_frag(wrow + kk + half * 8, wrow + kk + 16 + half * 8);
        acc = wmma_bf16(a, b, acc);
      }
      int n = nt * 16 + l16;
#pragma unroll
      for (int r = 0; r < 8; ++r) {
        int rrow = mt * 16 + r + half * 8;          // 0..47 within block
        int ge = eb * 8 + rrow / 6;                 // global edge
        float rg = sigmoidf_(acc[r] + Wrx[(size_t)ge * H + n]);
        float hn = (float)ldsA[rrow * 256 + n];
        ldsP[rrow * 256 + n] = rg * hn;             // disjoint slots, no atomics
      }
    }
  }
  __syncthreads();
  for (int p = threadIdx.x; p < 8 * 256; p += 256) {
    int e = p >> 8, n = p & 255;
    float s = 0.0f;
#pragma unroll
    for (int k = 0; k < 6; ++k) s += ldsP[(e * 6 + k) * 256 + n];
    sum_gh[(size_t)(eb * 8 + e) * H + n] = (__bf16)s;
  }
}

// ---------------------------------------------------------------------------
// Fused GRU update: two GEMMs (z and pre_h) + elementwise update, emask
// block = 16 edges; wave w -> n-tiles {w,w+8}
// ---------------------------------------------------------------------------
__global__ __launch_bounds__(256) void k_update(
    const __bf16* __restrict__ sumh_b, const __bf16* __restrict__ sumgh_b,
    const __bf16* __restrict__ Wz, const __bf16* __restrict__ Wh,
    const float* __restrict__ Axz, const float* __restrict__ Axh,
    const float* __restrict__ sumh_f, __bf16* __restrict__ hb) {
  int lane = threadIdx.x & 31;
  int wave = threadIdx.x >> 5;
  int half = lane >> 4, l16 = lane & 15;
  int mbase = blockIdx.x * 16;
  const __bf16* a1 = sumh_b + (size_t)(mbase + l16) * H;
  const __bf16* a2 = sumgh_b + (size_t)(mbase + l16) * H;
#pragma unroll
  for (int t = 0; t < 2; ++t) {
    int nt = wave + t * 8;
    const __bf16* wz = Wz + (size_t)(nt * 16 + l16) * H;
    const __bf16* wh = Wh + (size_t)(nt * 16 + l16) * H;
    v8f az = {}, ah = {};
    for (int kk = 0; kk < H; kk += 32) {
      v16bf f1 = make_frag(a1 + kk + half * 8, a1 + kk + 16 + half * 8);
      v16bf b1 = make_frag(wz + kk + half * 8, wz + kk + 16 + half * 8);
      az = wmma_bf16(f1, b1, az);
      v16bf f2 = make_frag(a2 + kk + half * 8, a2 + kk + 16 + half * 8);
      v16bf b2 = make_frag(wh + kk + half * 8, wh + kk + 16 + half * 8);
      ah = wmma_bf16(f2, b2, ah);
    }
    int n = nt * 16 + l16;
#pragma unroll
    for (int r = 0; r < 8; ++r) {
      int ge = mbase + r + half * 8;
      size_t off = (size_t)ge * H + n;
      float z = sigmoidf_(az[r] + Axz[off]);
      float ph = tanhf(ah[r] + Axh[off]);
      float hv = (1.0f - z) * sumh_f[off] + z * ph;
      if (ge == 0) hv = 0.0f;  // emask: edge 0 is padding
      hb[off] = (__bf16)hv;
    }
  }
}

// ---------------------------------------------------------------------------
// Readout: out = gelu(nei_msg @ Wo_h^T + Ax_o) * nmask   (exact erf GELU)
// ---------------------------------------------------------------------------
__global__ __launch_bounds__(256) void k_readout(
    const __bf16* __restrict__ nmb, const __bf16* __restrict__ Wo,
    const float* __restrict__ Axo, float* __restrict__ out) {
  int lane = threadIdx.x & 31;
  int wave = threadIdx.x >> 5;
  int half = lane >> 4, l16 = lane & 15;
  int mbase = blockIdx.x * 16;
  const __bf16* arow = nmb + (size_t)(mbase + l16) * H;
#pragma unroll
  for (int t = 0; t < 2; ++t) {
    int nt = wave + t * 8;
    const __bf16* wrow = Wo + (size_t)(nt * 16 + l16) * H;
    v8f acc = {};
    for (int kk = 0; kk < H; kk += 32) {
      v16bf a = make_frag(arow + kk + half * 8, arow + kk + 16 + half * 8);
      v16bf b = make_frag(wrow + kk + half * 8, wrow + kk + 16 + half * 8);
      acc = wmma_bf16(a, b, acc);
    }
    int n = nt * 16 + l16;
#pragma unroll
    for (int r = 0; r < 8; ++r) {
      int gn = mbase + r + half * 8;
      size_t off = (size_t)gn * H + n;
      float v = acc[r] + Axo[off];
      float g = 0.5f * v * (1.0f + erff(v * 0.70710678118654752f));
      if (gn == 0) g = 0.0f;  // nmask: node 0 is padding
      out[off] = g;
    }
  }
}

// ---------------------------------------------------------------------------
extern "C" void kernel_launch(void* const* d_in, const int* in_sizes, int n_in,
                              void* d_out, int out_size, void* d_ws,
                              size_t ws_size, hipStream_t stream) {
  (void)in_sizes; (void)n_in; (void)out_size; (void)ws_size;
  const float* fnode = (const float*)d_in[0];
  const float* fmess = (const float*)d_in[1];
  const int*   agraph = (const int*)d_in[2];
  const int*   bgraph = (const int*)d_in[3];
  const float* Wz_w = (const float*)d_in[4];
  const float* Wz_b = (const float*)d_in[5];
  const float* Wr_w = (const float*)d_in[6];
  const float* Ur_w = (const float*)d_in[7];
  const float* Ur_b = (const float*)d_in[8];
  const float* Wh_w = (const float*)d_in[9];
  const float* Wh_b = (const float*)d_in[10];
  const float* Wo_w = (const float*)d_in[11];
  const float* Wo_b = (const float*)d_in[12];

  size_t off = 0;
  auto alloc = [&](size_t bytes) -> char* {
    char* p = (char*)d_ws + off;
    off = (off + bytes + 255) & ~(size_t)255;
    return p;
  };
  __bf16* Wzx = (__bf16*)alloc(256 * 128 * 2);
  __bf16* Wzh = (__bf16*)alloc(256 * 256 * 2);
  __bf16* Whx = (__bf16*)alloc(256 * 128 * 2);
  __bf16* Whh = (__bf16*)alloc(256 * 256 * 2);
  __bf16* Wrb = (__bf16*)alloc(256 * 128 * 2);
  __bf16* Urb = (__bf16*)alloc(256 * 256 * 2);
  __bf16* Wox = (__bf16*)alloc(256 * 128 * 2);
  __bf16* Woh = (__bf16*)alloc(256 * 256 * 2);
  __bf16* xb   = (__bf16*)alloc((size_t)E_EDGES * 128 * 2);
  __bf16* fnb  = (__bf16*)alloc((size_t)N_NODES * 128 * 2);
  float*  Axz  = (float*)alloc((size_t)E_EDGES * H * 4);
  float*  Axh  = (float*)alloc((size_t)E_EDGES * H * 4);
  float*  Wrx  = (float*)alloc((size_t)E_EDGES * H * 4);
  float*  Axo  = (float*)alloc((size_t)N_NODES * H * 4);
  __bf16* hb      = (__bf16*)alloc((size_t)E_EDGES * H * 2);
  float*  sumh_f  = (float*)alloc((size_t)E_EDGES * H * 4);
  __bf16* sumh_b  = (__bf16*)alloc((size_t)E_EDGES * H * 2);
  __bf16* sumgh_b = (__bf16*)alloc((size_t)E_EDGES * H * 2);
  __bf16* nmb     = (__bf16*)alloc((size_t)N_NODES * H * 2);

  auto cw = [&](const float* src, __bf16* dst, int sc, int co, int cc, int dc) {
    int tot = 256 * dc;
    k_conv_w<<<(tot + 255) / 256, 256, 0, stream>>>(src, dst, 256, sc, co, cc, dc);
  };
  // W_z: [256, 367] -> x part (111 -> pad 128) + h part (256)
  cw(Wz_w, Wzx, 367, 0, 111, 128);
  cw(Wz_w, Wzh, 367, 111, 256, 256);
  cw(Wh_w, Whx, 367, 0, 111, 128);
  cw(Wh_w, Whh, 367, 111, 256, 256);
  cw(Wr_w, Wrb, 111, 0, 111, 128);
  cw(Ur_w, Urb, 256, 0, 256, 256);
  cw(Wo_w, Wox, 354, 0, 98, 128);
  cw(Wo_w, Woh, 354, 98, 256, 256);

  k_build_x<<<(E_EDGES * 128 + 255) / 256, 256, 0, stream>>>(fnode, fmess, xb);
  k_build_fn<<<(N_NODES * 128 + 255) / 256, 256, 0, stream>>>(fnode, fnb);
  k_zero_b<<<(E_EDGES * H + 255) / 256, 256, 0, stream>>>(hb, E_EDGES * H);

  // Loop-invariant precomputes (biases folded)
  k_gemm_bias<<<E_EDGES / 16, 256, 0, stream>>>(xb, Wzx, Wz_b, Axz, 128);
  k_gemm_bias<<<E_EDGES / 16, 256, 0, stream>>>(xb, Whx, Wh_b, Axh, 128);
  k_gemm_bias<<<E_EDGES / 16, 256, 0, stream>>>(xb, Wrb, Ur_b, Wrx, 128);
  k_gemm_bias<<<N_NODES / 16, 256, 0, stream>>>(fnb, Wox, Wo_b, Axo, 128);

  for (int d = 0; d < 4; ++d) {
    k_gather_sum<<<E_EDGES / 8, 256, 0, stream>>>(hb, bgraph, sumh_f, sumh_b,
                                                  E_EDGES);
    k_r_gemm<<<E_EDGES / 8, 256, 0, stream>>>(hb, bgraph, Urb, Wrx, sumgh_b);
    k_update<<<E_EDGES / 16, 256, 0, stream>>>(sumh_b, sumgh_b, Wzh, Whh, Axz,
                                               Axh, sumh_f, hb);
  }

  k_gather_sum<<<N_NODES / 8, 256, 0, stream>>>(hb, agraph, nullptr, nmb,
                                                N_NODES);
  k_readout<<<N_NODES / 16, 256, 0, stream>>>(nmb, Woh, Axo, (float*)d_out);
}
